// Encoder_overall_35888746725565
// MI455X (gfx1250) — matrix operand order repaired and checked by reading, not compiled, not run on hardware
//
#include <hip/hip_runtime.h>
#include <hip/hip_bf16.h>
#include <math.h>

// MI455X (gfx1250) implementation.
// Roofline: ~32 GFLOP of GEMM vs ~400MB traffic @ 23.3 TB/s -> matrix-core bound.
// bf16 WMMA (16x16x32) gives 8x MACs/instr vs f32 WMMA (16x16x4); bf16 adj
// (75.5 MB) + activations fit in the 192MB L2. Each wave computes a 32x32 tile
// (2x2 WMMAs) so each fragment is reused twice: 8 b128 loads -> 4 WMMAs per
// K-step (2 loads/WMMA), halving L2 traffic per FLOP vs a 16x16 tile.

typedef __bf16 bf16;
typedef __attribute__((ext_vector_type(8)))  bf16  v8bf;
typedef __attribute__((ext_vector_type(16))) bf16  v16bf;
typedef __attribute__((ext_vector_type(8)))  float v8f;

#define NROWS 6144
#define DIM1  3000
#define DIM2  1000
#define D1P   3008   // 3000 padded to mult of 32
#define D2P   1024   // 1000 padded to mult of 32
#define HD    128
#define LATD  64
#define NE    16

__device__ __forceinline__ unsigned short f2bf(float f) {
  unsigned u = __float_as_uint(f);
  unsigned r = u + 0x7FFFu + ((u >> 16) & 1u);   // round-to-nearest-even
  return (unsigned short)(r >> 16);
}

// ---------- conversion kernels ----------
// src f32 [R x C] -> dst bf16 [R x Cp], zero-padded columns
__global__ void k_cvt_pad(const float* __restrict__ src, unsigned short* __restrict__ dst,
                          int total, int C, int Cp) {
  int i = blockIdx.x * blockDim.x + threadIdx.x;
  if (i >= total) return;
  int r = i / Cp, c = i - r * Cp;
  dst[i] = (c < C) ? f2bf(src[(size_t)r * C + c]) : (unsigned short)0;
}

// src f32 [R x C] -> dst bf16 [Cp x Rp] (transposed), zero-padded
__global__ void k_cvt_tr(const float* __restrict__ src, unsigned short* __restrict__ dst,
                         int total, int R, int C, int Rp) {
  int i = blockIdx.x * blockDim.x + threadIdx.x;
  if (i >= total) return;
  int c = i / Rp, r = i - c * Rp;
  dst[i] = (r < R && c < C) ? f2bf(src[(size_t)r * C + c]) : (unsigned short)0;
}

// ---------- WMMA fragment loads (CDNA5 ISA 7.12.2 layouts) ----------
//  A 16x32 bf16 : lane(0-15)=row, half selects K+0..7/+16..23 vs K+8..15/+24..31
//  B 32x16 bf16 : lane(0-15)=col, half selects K+0..7/+16..23 vs K+8..15/+24..31
__device__ __forceinline__ v16bf ldfragA(const unsigned short* p) {
  v8bf lo = *(const v8bf*)(const void*)p;
  v8bf hi = *(const v8bf*)(const void*)(p + 16);
  return __builtin_shufflevector(lo, hi, 0,1,2,3,4,5,6,7,8,9,10,11,12,13,14,15);
}
__device__ __forceinline__ v16bf ldfragB(const unsigned short* p) {
  v8bf lo = *(const v8bf*)(const void*)p;
  v8bf hi = *(const v8bf*)(const void*)(p + 16);
  return __builtin_shufflevector(lo, hi, 0,1,2,3,4,5,6,7,8,9,10,11,12,13,14,15);
}
__device__ __forceinline__ void storeCD(float* __restrict__ C, const v8f& acc,
                                        int mb, int n, int ldc, int nvalid,
                                        const float* __restrict__ bias) {
  if (n >= nvalid) return;
  float b = bias ? bias[n] : 0.0f;
#pragma unroll
  for (int r = 0; r < 8; ++r)
    C[(size_t)(mb + r) * ldc + n] = acc[r] + b;
}

// ---------- WMMA GEMM: C[M x nvalid] (f32,+bias) = A[M x K]bf16 * Bt[Np x K]bf16^T ----------
// 4 waves / block, one 32x32 tile per wave (2x2 of 16x16 WMMAs).
__global__ __launch_bounds__(128)
void k_gemm_bf16(const unsigned short* __restrict__ A,
                 const unsigned short* __restrict__ Bt,
                 float* __restrict__ C,
                 int K, int ntn32, int ntiles, int ldc, int nvalid,
                 const float* __restrict__ bias) {
  const int tile = blockIdx.x * 4 + (threadIdx.x >> 5);
  if (tile >= ntiles) return;
  const int tm = tile / ntn32, tn = tile - tm * ntn32;
  const int lane = threadIdx.x & 31;
  const int half = lane >> 4, r16 = lane & 15;

  const unsigned short* a0 = A  + (size_t)(tm * 32 + r16) * K + half * 8;
  const unsigned short* a1 = a0 + (size_t)16 * K;
  const unsigned short* b0 = Bt + (size_t)(tn * 32 + r16) * K + half * 8;
  const unsigned short* b1 = b0 + (size_t)16 * K;

  v8f c00, c01, c10, c11;
#pragma unroll
  for (int i = 0; i < 8; ++i) { c00[i] = 0.f; c01[i] = 0.f; c10[i] = 0.f; c11[i] = 0.f; }

  for (int k0 = 0; k0 < K; k0 += 32) {
    __builtin_prefetch(a0 + k0 + 512, 0, 1);   // global_prefetch_b8
    __builtin_prefetch(a1 + k0 + 512, 0, 1);
    __builtin_prefetch(b0 + k0 + 512, 0, 1);
    __builtin_prefetch(b1 + k0 + 512, 0, 1);
    v16bf av0 = ldfragA(a0 + k0);
    v16bf av1 = ldfragA(a1 + k0);
    v16bf bv0 = ldfragB(b0 + k0);
    v16bf bv1 = ldfragB(b1 + k0);
    c00 = __builtin_amdgcn_wmma_f32_16x16x32_bf16(false, av0, false, bv0, (short)0, c00, false, false);
    c01 = __builtin_amdgcn_wmma_f32_16x16x32_bf16(false, av0, false, bv1, (short)0, c01, false, false);
    c10 = __builtin_amdgcn_wmma_f32_16x16x32_bf16(false, av1, false, bv0, (short)0, c10, false, false);
    c11 = __builtin_amdgcn_wmma_f32_16x16x32_bf16(false, av1, false, bv1, (short)0, c11, false, false);
  }

  const int mb0 = tm * 32 + half * 8;
  const int n0  = tn * 32 + r16;
  storeCD(C, c00, mb0,      n0,      ldc, nvalid, bias);
  storeCD(C, c01, mb0,      n0 + 16, ldc, nvalid, bias);
  storeCD(C, c10, mb0 + 16, n0,      ldc, nvalid, bias);
  storeCD(C, c11, mb0 + 16, n0 + 16, ldc, nvalid, bias);
}

// ---------- BatchNorm ----------
__global__ __launch_bounds__(256)
void k_bn_stats(const float* __restrict__ H, int N, int Hdim,
                float* __restrict__ mean, float* __restrict__ var) {
  __shared__ float ss[256], sq[256];
  const int h = blockIdx.x;
  float s = 0.f, q = 0.f;
  for (int r = threadIdx.x; r < N; r += 256) {
    float v = H[(size_t)r * Hdim + h];
    s += v; q += v * v;
  }
  ss[threadIdx.x] = s; sq[threadIdx.x] = q;
  __syncthreads();
  for (int w = 128; w > 0; w >>= 1) {
    if (threadIdx.x < w) { ss[threadIdx.x] += ss[threadIdx.x + w]; sq[threadIdx.x] += sq[threadIdx.x + w]; }
    __syncthreads();
  }
  if (threadIdx.x == 0) {
    float m = ss[0] / N;
    mean[h] = m;
    var[h] = sq[0] / N - m * m;
  }
}

__global__ void k_bn_apply(const float* __restrict__ H, const float* __restrict__ mean,
                           const float* __restrict__ var, const float* __restrict__ gamma,
                           const float* __restrict__ beta, unsigned short* __restrict__ out,
                           int total, int Hdim) {
  int i = blockIdx.x * blockDim.x + threadIdx.x;
  if (i >= total) return;
  int h = i % Hdim;
  float v = (H[i] - mean[h]) * rsqrtf(var[h] + 1e-5f) * gamma[h] + beta[h];
  out[i] = f2bf(v);
}

// ---------- Product of Experts ----------
__global__ void k_poe(const float* __restrict__ mu1, const float* __restrict__ lv1,
                      const float* __restrict__ mu2, const float* __restrict__ lv2,
                      float* __restrict__ pdmu, float* __restrict__ pdlv,
                      unsigned short* __restrict__ zb, int total) {
  int i = blockIdx.x * blockDim.x + threadIdx.x;
  if (i >= total) return;
  float v1 = __expf(lv1[i]) + 1e-8f, v2 = __expf(lv2[i]) + 1e-8f;
  float T1 = 1.f / (v1 + 1e-8f), T2 = 1.f / (v2 + 1e-8f);
  float Ts = T1 + T2;
  float m = (mu1[i] * T1 + mu2[i] * T2) / Ts;
  pdmu[i] = m;
  pdlv[i] = __logf(1.f / Ts + 1e-8f);
  zb[i] = f2bf(m);
}

// ---------- MoE noisy-top-k gating ----------
__device__ __forceinline__ unsigned pcg(unsigned v) {
  unsigned s = v * 747796405u + 2891336453u;
  unsigned w = ((s >> ((s >> 28) + 4u)) ^ s) * 277803737u;
  return (w >> 22) ^ w;
}
__device__ __forceinline__ float gaussn(unsigned idx) {
  unsigned a = pcg(idx * 2u + 1u), b = pcg(idx * 2u + 2u);
  float u1 = ((float)((a & 0xFFFFFFu) + 1u)) * (1.0f / 16777218.0f);  // (0,1]
  float u2 = ((float)(b & 0xFFFFFFu)) * (1.0f / 16777216.0f);
  return sqrtf(-2.0f * __logf(u1)) * __cosf(6.2831853f * u2);
}
__device__ __forceinline__ float phi(float x) { return 0.5f * erfcf(-x * 0.70710678118f); }

__global__ void k_zero(float* p, int n) {
  int i = blockIdx.x * blockDim.x + threadIdx.x;
  if (i < n) p[i] = 0.0f;
}

__global__ __launch_bounds__(64)
void k_gate(const float* __restrict__ z, const float* __restrict__ wg,
            const float* __restrict__ wn, float* __restrict__ gates,
            float* __restrict__ acc, int N) {
  int n = blockIdx.x * blockDim.x + threadIdx.x;
  if (n >= N) return;
  const float* zr = z + (size_t)n * LATD;
  float clean[NE], sd[NE], noisy[NE];
#pragma unroll
  for (int e = 0; e < NE; ++e) {
    float cg = 0.f, cn = 0.f;
    for (int l = 0; l < LATD; ++l) {
      float zv = zr[l];
      cg += zv * wg[l * NE + e];
      cn += zv * wn[l * NE + e];
    }
    clean[e] = cg;
    float sp = (cn > 20.f) ? cn : log1pf(__expf(cn));   // softplus
    sd[e] = sp + 0.01f;
    noisy[e] = cg + gaussn((unsigned)(n * NE + e)) * sd[e];
  }
  // top-3 selection
  float v0 = -1e30f, v1 = -1e30f, v2 = -1e30f;
  int i0 = 0, i1 = 0;
#pragma unroll
  for (int e = 0; e < NE; ++e) {
    float v = noisy[e];
    if (v > v0)      { v2 = v1; v1 = v0; i1 = i0; v0 = v; i0 = e; }
    else if (v > v1) { v2 = v1; v1 = v; i1 = e; }
    else if (v > v2) { v2 = v; }
  }
  float e1 = __expf(v1 - v0);
  float inv = 1.f / (1.0f + e1);
  float g0 = inv, g1 = e1 * inv;
  float* gr = gates + (size_t)n * NE;
#pragma unroll
  for (int e = 0; e < NE; ++e) gr[e] = 0.0f;
  gr[i0] = g0; gr[i1] = g1;
  atomicAdd(&acc[i0], g0);
  atomicAdd(&acc[i1], g1);
#pragma unroll
  for (int e = 0; e < NE; ++e) {
    float thr = (noisy[e] > v2) ? v2 : v1;   // thr_in : thr_out
    atomicAdd(&acc[NE + e], phi((clean[e] - thr) / sd[e]));
  }
}

__global__ void k_gate_loss(const float* __restrict__ acc, float* __restrict__ out) {
  if (threadIdx.x != 0 || blockIdx.x != 0) return;
  float loss = 0.f;
  for (int p = 0; p < 2; ++p) {
    const float* a = acc + p * NE;
    float s = 0.f;
    for (int e = 0; e < NE; ++e) s += a[e];
    float m = s / NE, vq = 0.f;
    for (int e = 0; e < NE; ++e) { float d = a[e] - m; vq += d * d; }
    float var = vq / (NE - 1);
    loss += var / (m * m + 1e-10f);
  }
  out[0] = loss;
}

// ---------- host-side orchestration ----------
extern "C" void kernel_launch(void* const* d_in, const int* in_sizes, int n_in,
                              void* d_out, int out_size, void* d_ws, size_t ws_size,
                              hipStream_t stream) {
  const float* x1     = (const float*)d_in[0];
  const float* x2     = (const float*)d_in[1];
  const float* adj    = (const float*)d_in[2];
  const float* W1     = (const float*)d_in[3];
  const float* gamma1 = (const float*)d_in[4];
  const float* beta1  = (const float*)d_in[5];
  const float* Wmu1   = (const float*)d_in[6];
  const float* bmu1   = (const float*)d_in[7];
  const float* Wlv1   = (const float*)d_in[8];
  const float* blv1   = (const float*)d_in[9];
  const float* W2     = (const float*)d_in[10];
  const float* gamma2 = (const float*)d_in[11];
  const float* beta2  = (const float*)d_in[12];
  const float* Wmu2   = (const float*)d_in[13];
  const float* bmu2   = (const float*)d_in[14];
  const float* Wlv2   = (const float*)d_in[15];
  const float* blv2   = (const float*)d_in[16];
  const float* w_gate = (const float*)d_in[17];
  const float* w_noise= (const float*)d_in[18];
  const float* Wd1    = (const float*)d_in[19];
  const float* Wfc1   = (const float*)d_in[20];
  const float* bfc1   = (const float*)d_in[21];
  const float* Wd2    = (const float*)d_in[22];
  const float* Wfc2   = (const float*)d_in[23];
  const float* bfc2   = (const float*)d_in[24];
  (void)in_sizes; (void)n_in; (void)out_size; (void)ws_size;

  // output layout: recon1, recon2, pd_mu, pd_logvar, gates, moe_loss
  float* out      = (float*)d_out;
  float* o_recon1 = out;
  float* o_recon2 = o_recon1 + (size_t)NROWS * DIM1;
  float* o_pdmu   = o_recon2 + (size_t)NROWS * DIM2;
  float* o_pdlv   = o_pdmu   + (size_t)NROWS * LATD;
  float* o_gates  = o_pdlv   + (size_t)NROWS * LATD;
  float* o_loss   = o_gates  + (size_t)NROWS * NE;

  // workspace bump allocator (256B aligned)
  char* wsb = (char*)d_ws;
  size_t off = 0;
  auto alloc = [&](size_t bytes) -> void* {
    off = (off + 255) & ~(size_t)255;
    void* p = wsb + off; off += bytes; return p;
  };
  unsigned short* adjb  = (unsigned short*)alloc((size_t)NROWS * NROWS * 2);
  unsigned short* x1b   = (unsigned short*)alloc((size_t)NROWS * D1P * 2);
  unsigned short* x2b   = (unsigned short*)alloc((size_t)NROWS * D2P * 2);
  unsigned short* w1t   = (unsigned short*)alloc((size_t)HD * D1P * 2);
  unsigned short* w2t   = (unsigned short*)alloc((size_t)HD * D2P * 2);
  unsigned short* wmu1t = (unsigned short*)alloc((size_t)LATD * HD * 2);
  unsigned short* wlv1t = (unsigned short*)alloc((size_t)LATD * HD * 2);
  unsigned short* wmu2t = (unsigned short*)alloc((size_t)LATD * HD * 2);
  unsigned short* wlv2t = (unsigned short*)alloc((size_t)LATD * HD * 2);
  unsigned short* wd1t  = (unsigned short*)alloc((size_t)LATD * LATD * 2);
  unsigned short* wd2t  = (unsigned short*)alloc((size_t)LATD * LATD * 2);
  unsigned short* wfc1t = (unsigned short*)alloc((size_t)D1P * LATD * 2);
  unsigned short* wfc2t = (unsigned short*)alloc((size_t)D2P * LATD * 2);
  float* t1f  = (float*)alloc((size_t)NROWS * HD * 4);
  float* t2f  = (float*)alloc((size_t)NROWS * HD * 4);
  unsigned short* t1t = (unsigned short*)alloc((size_t)HD * NROWS * 2);
  unsigned short* t2t = (unsigned short*)alloc((size_t)HD * NROWS * 2);
  float* h1f  = (float*)alloc((size_t)NROWS * HD * 4);
  float* h2f  = (float*)alloc((size_t)NROWS * HD * 4);
  unsigned short* h1n = (unsigned short*)alloc((size_t)NROWS * HD * 2);
  unsigned short* h2n = (unsigned short*)alloc((size_t)NROWS * HD * 2);
  float* mu1f = (float*)alloc((size_t)NROWS * LATD * 4);
  float* lv1f = (float*)alloc((size_t)NROWS * LATD * 4);
  float* mu2f = (float*)alloc((size_t)NROWS * LATD * 4);
  float* lv2f = (float*)alloc((size_t)NROWS * LATD * 4);
  unsigned short* zb = (unsigned short*)alloc((size_t)NROWS * LATD * 2);
  float* zd1f = (float*)alloc((size_t)NROWS * LATD * 4);
  float* zd2f = (float*)alloc((size_t)NROWS * LATD * 4);
  unsigned short* zd1t = (unsigned short*)alloc((size_t)LATD * NROWS * 2);
  unsigned short* zd2t = (unsigned short*)alloc((size_t)LATD * NROWS * 2);
  float* a1f  = (float*)alloc((size_t)NROWS * LATD * 4);
  float* a2f  = (float*)alloc((size_t)NROWS * LATD * 4);
  unsigned short* a1b = (unsigned short*)alloc((size_t)NROWS * LATD * 2);
  unsigned short* a2b = (unsigned short*)alloc((size_t)NROWS * LATD * 2);
  float* stats = (float*)alloc(4 * HD * 4);   // mean1,var1,mean2,var2
  float* gacc  = (float*)alloc(2 * NE * 4);   // importance, load
  float* mean1 = stats, *var1 = stats + HD, *mean2 = stats + 2 * HD, *var2 = stats + 3 * HD;

  auto cvtPad = [&](const float* src, unsigned short* dst, int R, int C, int Cp) {
    int total = R * Cp;
    k_cvt_pad<<<(total + 255) / 256, 256, 0, stream>>>(src, dst, total, C, Cp);
  };
  auto cvtTr = [&](const float* src, unsigned short* dst, int R, int C, int Cp, int Rp) {
    int total = Cp * Rp;
    k_cvt_tr<<<(total + 255) / 256, 256, 0, stream>>>(src, dst, total, R, C, Rp);
  };
  auto gemm = [&](const unsigned short* A, const unsigned short* Bt, float* Cc,
                  int M, int K, int Np, int ldc, int nvalid, const float* bias) {
    int ntn32 = Np / 32;
    int ntiles = (M / 32) * ntn32;
    k_gemm_bf16<<<(ntiles + 3) / 4, 128, 0, stream>>>(A, Bt, Cc, K, ntn32, ntiles,
                                                      ldc, nvalid, bias);
  };

  // --- convert/transposes ---
  cvtPad(adj, adjb, NROWS, NROWS, NROWS);
  cvtPad(x1,  x1b,  NROWS, DIM1, D1P);
  cvtPad(x2,  x2b,  NROWS, DIM2, D2P);
  cvtTr(W1,   w1t,   DIM1, HD,   HD,   D1P);
  cvtTr(W2,   w2t,   DIM2, HD,   HD,   D2P);
  cvtTr(Wmu1, wmu1t, HD,   LATD, LATD, HD);
  cvtTr(Wlv1, wlv1t, HD,   LATD, LATD, HD);
  cvtTr(Wmu2, wmu2t, HD,   LATD, LATD, HD);
  cvtTr(Wlv2, wlv2t, HD,   LATD, LATD, HD);
  cvtTr(Wd1,  wd1t,  LATD, LATD, LATD, LATD);
  cvtTr(Wd2,  wd2t,  LATD, LATD, LATD, LATD);
  cvtTr(Wfc1, wfc1t, LATD, DIM1, D1P,  LATD);
  cvtTr(Wfc2, wfc2t, LATD, DIM2, D2P,  LATD);

  // --- encoder modality 1 ---
  gemm(x1b, w1t, t1f, NROWS, D1P, HD, HD, HD, nullptr);           // T1 = x1 @ W1
  cvtTr(t1f, t1t, NROWS, HD, HD, NROWS);
  gemm(adjb, t1t, h1f, NROWS, NROWS, HD, HD, HD, nullptr);        // H1 = adj @ T1
  k_bn_stats<<<HD, 256, 0, stream>>>(h1f, NROWS, HD, mean1, var1);
  k_bn_apply<<<(NROWS * HD + 255) / 256, 256, 0, stream>>>(h1f, mean1, var1, gamma1, beta1,
                                                           h1n, NROWS * HD, HD);
  gemm(h1n, wmu1t, mu1f, NROWS, HD, LATD, LATD, LATD, bmu1);      // mu1
  gemm(h1n, wlv1t, lv1f, NROWS, HD, LATD, LATD, LATD, blv1);      // logvar1

  // --- encoder modality 2 ---
  gemm(x2b, w2t, t2f, NROWS, D2P, HD, HD, HD, nullptr);
  cvtTr(t2f, t2t, NROWS, HD, HD, NROWS);
  gemm(adjb, t2t, h2f, NROWS, NROWS, HD, HD, HD, nullptr);
  k_bn_stats<<<HD, 256, 0, stream>>>(h2f, NROWS, HD, mean2, var2);
  k_bn_apply<<<(NROWS * HD + 255) / 256, 256, 0, stream>>>(h2f, mean2, var2, gamma2, beta2,
                                                           h2n, NROWS * HD, HD);
  gemm(h2n, wmu2t, mu2f, NROWS, HD, LATD, LATD, LATD, bmu2);
  gemm(h2n, wlv2t, lv2f, NROWS, HD, LATD, LATD, LATD, blv2);

  // --- Product of Experts -> pd_mu, pd_logvar, z(bf16) ---
  k_poe<<<(NROWS * LATD + 255) / 256, 256, 0, stream>>>(mu1f, lv1f, mu2f, lv2f,
                                                        o_pdmu, o_pdlv, zb, NROWS * LATD);

  // --- MoE gating ---
  k_zero<<<1, 32, 0, stream>>>(gacc, 2 * NE);
  k_gate<<<(NROWS + 63) / 64, 64, 0, stream>>>(o_pdmu, w_gate, w_noise, o_gates, gacc, NROWS);
  k_gate_loss<<<1, 32, 0, stream>>>(gacc, o_loss);

  // --- decoder modality 1 ---
  gemm(zb, wd1t, zd1f, NROWS, LATD, LATD, LATD, LATD, nullptr);   // zWd1
  cvtTr(zd1f, zd1t, NROWS, LATD, LATD, NROWS);
  gemm(adjb, zd1t, a1f, NROWS, NROWS, LATD, LATD, LATD, nullptr); // adj @ zWd1
  cvtPad(a1f, a1b, NROWS, LATD, LATD);
  gemm(a1b, wfc1t, o_recon1, NROWS, LATD, D1P, DIM1, DIM1, bfc1); // recon1

  // --- decoder modality 2 ---
  gemm(zb, wd2t, zd2f, NROWS, LATD, LATD, LATD, LATD, nullptr);
  cvtTr(zd2f, zd2t, NROWS, LATD, LATD, NROWS);
  gemm(adjb, zd2t, a2f, NROWS, NROWS, LATD, LATD, LATD, nullptr);
  cvtPad(a2f, a2b, NROWS, LATD, LATD);
  gemm(a2b, wfc2t, o_recon2, NROWS, LATD, D2P, DIM2, DIM2, bfc2);
}
